// CrossScaleAttention_13572096656005
// MI455X (gfx1250) — matrix operand order repaired and checked by reading, not compiled, not run on hardware
//
#include <hip/hip_runtime.h>

// ---------------------------------------------------------------------------
// CrossScaleAttention for MI455X (gfx1250), wave32 + WMMA + TDM.
// Pipeline (launch order; k_tconv defined first for disasm visibility):
//   1. k_embed_match : fused conv1x1 (Wa 64ch + W1 32ch) via WMMA, PReLU -> f16
//   2. k_pool_ref    : 2x2 mean pool + conv1x1 W2 + PReLU -> ref f32
//   3. k_wn          : 3x3 patches of ref, L2-normalized -> wn f16 [L,288]
//   4. k_xp          : 3x3 patches of match_in -> xp f16 [Q,288]
//   5. k_rawW        : 6x6/stride-2 patches of embed -> rawW f16 [L, 64*36]
//   6. k_attn        : scores = xp @ wn^T  (WMMA GEMM, K=288) -> f32 [Q,L]
//                      B-tiles double-buffered into LDS by the TDM
//   7. k_softmax     : per-q max/exp/sum over L; stores exp() f16, rowsum f32
//   8. k_tconv       : P = exp @ rawW (WMMA GEMM, K=2304), epilogue * 1/rowsum
//                      A- and B-tiles double-buffered into LDS by the TDM
//   9. k_yt          : overlap-add of 6x6 patches (gather form, 9 taps)
//  10. k_down        : 3x3 stride-2 conv + bias, * 1/6 -> d_out
// ---------------------------------------------------------------------------

typedef _Float16 f16;
typedef __attribute__((ext_vector_type(16))) _Float16     h16;
typedef __attribute__((ext_vector_type(8)))  float        f32x8;
typedef __attribute__((ext_vector_type(4)))  unsigned int u32x4;
typedef __attribute__((ext_vector_type(8)))  int          i32x8;
typedef __attribute__((ext_vector_type(4)))  int          i32x4;

#define BB    2
#define CC    64
#define HH    96
#define HWp   (96 * 96)     // 9216 queries / pixels
#define LL    2304          // 48*48 reference patches
#define K288  288           // 32 * 3 * 3
#define NDYN  2304          // 64 * 36 dynamic-kernel taps

// ---- TDM availability / arity detection (probe-documented) ----------------
#if defined(__has_builtin)
#  if __has_builtin(__builtin_amdgcn_tensor_load_to_lds)
#    define HAVE_TDM 1
#  endif
#endif
#ifndef HAVE_TDM
#  define HAVE_TDM 0
#endif
#if __has_include(<hip/amd_detail/amd_gfx1250_TDM.h>)
#  define TDM_SIX_ARGS 1    // therock-10.0 headers -> clang-23 6-arg builtin
#else
#  define TDM_SIX_ARGS 0    // ROCm 7.2 clang-22 5-arg builtin
#endif

static __device__ __forceinline__ f32x8 wmma_f16(h16 a, h16 b, f32x8 c) {
  return __builtin_amdgcn_wmma_f32_16x16x32_f16(
      /*neg_a=*/false, a, /*neg_b=*/false, b,
      /*c_mod=*/(short)0, c, /*reuse_a=*/false, /*reuse_b=*/false);
}

#if HAVE_TDM
// Issue a TDM 2D-tile load of f16 data: tile (rows x width) from a row-major
// tensor with leading dimension ld (elements), into LDS at lds_off.
// D# packed per CDNA5 ISA ch.8 (count=1, type=2, data_size=2B).
static __device__ __forceinline__ void
tdm_load_tile_f16(unsigned lds_off, const f16* gptr, int width, int rows,
                  int ld, int nrows_total) {
  unsigned long long ga = (unsigned long long)(size_t)gptr;
  u32x4 g0 = {1u,                                    // count=1
              lds_off,                               // lds_addr (bytes)
              (unsigned)ga,                          // global_addr lo32
              (unsigned)((ga >> 32) & 0x1FFFFFFull) | (2u << 30)};  // hi | type=2
  unsigned long long seg[4];
  seg[0] = (1ull << 16)                                       // data_size = 2B
         | ((unsigned long long)(ld & 0xFFFF) << 48);         // tensor_dim0 lo16
  seg[1] = (unsigned long long)((unsigned)ld >> 16)           // tensor_dim0 hi16
         | ((unsigned long long)(unsigned)nrows_total << 16)  // tensor_dim1
         | ((unsigned long long)(unsigned)width << 48);       // tile_dim0
  seg[2] = (unsigned long long)(unsigned)rows                 // tile_dim1
         | ((unsigned long long)(unsigned)ld << 32);          // dim0 stride lo32
  seg[3] = 0ull;
  i32x8 g1;
  __builtin_memcpy(&g1, seg, 32);
  i32x4 z4 = {0, 0, 0, 0};
#if TDM_SIX_ARGS
  i32x8 z8 = {0, 0, 0, 0, 0, 0, 0, 0};
  __builtin_amdgcn_tensor_load_to_lds(g0, g1, z4, z4, z8, 0);
#else
  __builtin_amdgcn_tensor_load_to_lds(g0, g1, z4, z4, 0);
#endif
}
#endif

// ---------------------------------------------------------------------------
// 8. Dynamic-kernel GEMM: P[q,n] = (sum_l e[q,l]*rawW[l,n]) / rowsum[q]
//    K=2304, 64(M) x 128(N) block tile, 4 waves. Both the A tile (64x32 of
//    eprob) and the B tile (32x128 of rawW) are double-buffered into LDS by
//    the Tensor Data Mover; the inner loop is pure ds_load + v_wmma.
// ---------------------------------------------------------------------------
__global__ __launch_bounds__(128)
void k_tconv(const f16* __restrict__ eprob, const f16* __restrict__ rawW,
             const float* __restrict__ rowsum, float* __restrict__ P) {
  const int b  = blockIdx.z;
  const int q0 = blockIdx.x * 64;
  const int n0 = blockIdx.y * 128;
  __shared__ __align__(32) f16 lbB[2][32 * 128];  // [buf][k_local][n_local]
  __shared__ __align__(32) f16 lbA[2][64 * 32];   // [buf][m_local][k_local]
  const int tid  = threadIdx.x, wv = tid >> 5, lane = tid & 31;
  const int lrow = lane & 15, lhi = lane >> 4;
  const f16* eb = eprob + (size_t)b * HWp * LL;
  const f16* rb = rawW + (size_t)b * LL * NDYN;
  f32x8 acc[8] = {};
  const int STEPS = LL / 32;   // 72

#if HAVE_TDM
  const unsigned ldsB0 = (unsigned)(size_t)(&lbB[0][0]);
  const unsigned ldsB1 = (unsigned)(size_t)(&lbB[1][0]);
  const unsigned ldsA0 = (unsigned)(size_t)(&lbA[0][0]);
  const unsigned ldsA1 = (unsigned)(size_t)(&lbA[1][0]);
  if (tid < 32) {  // wave 0 drives the TDM (EXEC-independent wave op)
    tdm_load_tile_f16(ldsB0, rb + n0, 128, 32, NDYN, LL);
    tdm_load_tile_f16(ldsA0, eb + (size_t)q0 * LL, 32, 64, LL, HWp);
  }
#endif

  for (int it = 0; it < STEPS; ++it) {
    const int k0 = it * 32;
    __syncthreads();   // prior reads of the buffers about to be overwritten
#if HAVE_TDM
    if (tid < 32) {
      if (it + 1 < STEPS) {
        tdm_load_tile_f16((it & 1) ? ldsB0 : ldsB1,
                          rb + (size_t)(k0 + 32) * NDYN + n0, 128, 32, NDYN,
                          LL);
        tdm_load_tile_f16((it & 1) ? ldsA0 : ldsA1,
                          eb + (size_t)q0 * LL + (k0 + 32), 32, 64, LL, HWp);
        // TDM ops complete in-order: <=2 outstanding => tiles `it` resident.
        __builtin_amdgcn_s_wait_tensorcnt(2);
      } else {
        __builtin_amdgcn_s_wait_tensorcnt(0);
      }
    }
#else
    {
      int kk = tid >> 2, nc = (tid & 3) * 32;
      *(h16*)&lbB[it & 1][kk * 128 + nc +  0] =
          *(const h16*)&rb[(size_t)(k0 + kk) * NDYN + n0 + nc +  0];
      *(h16*)&lbB[it & 1][kk * 128 + nc + 16] =
          *(const h16*)&rb[(size_t)(k0 + kk) * NDYN + n0 + nc + 16];
      int mm = tid >> 1, kc = (tid & 1) * 16;
      *(h16*)&lbA[it & 1][mm * 32 + kc] =
          *(const h16*)&eb[(size_t)(q0 + mm) * LL + k0 + kc];
    }
#endif
    __syncthreads();
    const f16* lbufB = lbB[it & 1];
    const f16* lbufA = lbA[it & 1];
    h16 afrag = *(const h16*)&lbufA[(16 * wv + lrow) * 32 + lhi * 16];
#pragma unroll
    for (int nt = 0; nt < 8; ++nt) {
      h16 bfrag;
#pragma unroll
      for (int s = 0; s < 16; ++s)
        bfrag[s] = lbufB[(lhi * 16 + s) * 128 + nt * 16 + lrow];
      acc[nt] = wmma_f16(afrag, bfrag, acc[nt]);
    }
  }

  const float* rs = rowsum + (size_t)b * HWp;
  float* Pb = P + (size_t)b * HWp * (size_t)NDYN;
#pragma unroll
  for (int nt = 0; nt < 8; ++nt) {
#pragma unroll
    for (int v = 0; v < 8; ++v) {
      int q = q0 + 16 * wv + v + 8 * lhi;
      int n = n0 + nt * 16 + lrow;
      Pb[(size_t)q * NDYN + n] = acc[nt][v] / rs[q];
    }
  }
}

// ---------------------------------------------------------------------------
// 6. Attention GEMM: scores[q,l] = sum_c xp[q,c] * wn[l,c]   (K=288)
//    64(M) x 128(N) block tile, 4 waves; wn tile (128 rows x 32 cols, i.e.
//    column-major B) double-buffered into LDS by the Tensor Data Mover.
// ---------------------------------------------------------------------------
__global__ __launch_bounds__(128)
void k_attn(const f16* __restrict__ xp, const f16* __restrict__ wn,
            float* __restrict__ scores) {
  const int b  = blockIdx.z;
  const int q0 = blockIdx.x * 64;
  const int l0 = blockIdx.y * 128;
  __shared__ __align__(32) f16 lb[2][128 * 32];   // [buf][n_local][k_local]
  const int tid  = threadIdx.x, wv = tid >> 5, lane = tid & 31;
  const int lrow = lane & 15, lhi = lane >> 4;
  const f16* xpb = xp + (size_t)b * HWp * K288;
  const f16* wnb = wn + (size_t)b * LL * K288;
  f32x8 acc[8] = {};
  const int STEPS = K288 / 32;   // 9

#if HAVE_TDM
  const unsigned lds0 = (unsigned)(size_t)(&lb[0][0]);
  const unsigned lds1 = (unsigned)(size_t)(&lb[1][0]);
  if (tid < 32)
    tdm_load_tile_f16(lds0, wnb + (size_t)l0 * K288 + 0, 32, 128, K288, LL);
#endif

  for (int it = 0; it < STEPS; ++it) {
    const int k0 = it * 32;
    __syncthreads();
#if HAVE_TDM
    if (tid < 32) {
      if (it + 1 < STEPS) {
        tdm_load_tile_f16((it & 1) ? lds0 : lds1,
                          wnb + (size_t)l0 * K288 + (k0 + 32), 32, 128, K288,
                          LL);
        __builtin_amdgcn_s_wait_tensorcnt(1);  // tile `it` resident (in-order)
      } else {
        __builtin_amdgcn_s_wait_tensorcnt(0);
      }
    }
#else
    {
      // cooperative fallback: 128 rows x 32 halves
      *(h16*)&lb[it & 1][tid * 32 +  0] =
          *(const h16*)&wnb[(size_t)(l0 + tid) * K288 + k0 +  0];
      *(h16*)&lb[it & 1][tid * 32 + 16] =
          *(const h16*)&wnb[(size_t)(l0 + tid) * K288 + k0 + 16];
    }
#endif
    __syncthreads();
    const f16* lbuf = lb[it & 1];
    h16 afrag =
        *(const h16*)&xpb[(size_t)(q0 + 16 * wv + lrow) * K288 + k0 + lhi * 16];
#pragma unroll
    for (int nt = 0; nt < 8; ++nt) {
      h16 bfrag = *(const h16*)&lbuf[(nt * 16 + lrow) * 32 + lhi * 16];
      acc[nt] = wmma_f16(afrag, bfrag, acc[nt]);
    }
    if (it + 1 < STEPS)
      __builtin_prefetch(
          &xpb[(size_t)(q0 + 16 * wv + lrow) * K288 + k0 + 32 + lhi * 16], 0, 1);
  }

  float* sb = scores + (size_t)b * HWp * LL;
#pragma unroll
  for (int nt = 0; nt < 8; ++nt) {
#pragma unroll
    for (int v = 0; v < 8; ++v) {
      int q = q0 + 16 * wv + v + 8 * lhi;
      int l = l0 + nt * 16 + lrow;
      sb[(size_t)q * LL + l] = acc[nt][v];
    }
  }
}

// ---------------------------------------------------------------------------
// 1. Fused conv1x1: [pixels x 64] * [64 x 96] -> embed(64) | match(32), PReLU
// ---------------------------------------------------------------------------
__global__ __launch_bounds__(128)
void k_embed_match(const float* __restrict__ x,
                   const float* __restrict__ Wa, const float* __restrict__ ba,
                   const float* __restrict__ aa,
                   const float* __restrict__ W1, const float* __restrict__ b1,
                   const float* __restrict__ a1,
                   f16* __restrict__ embed, f16* __restrict__ match) {
  const int b  = blockIdx.y;
  const int p0 = blockIdx.x * 64;
  __shared__ __align__(32) f16 sA[64 * 64];  // [pixel][cin]
  __shared__ __align__(32) f16 sW[96 * 64];  // [nout][cin]
  const int tid = threadIdx.x;

  const float* xb = x + (size_t)b * CC * HWp;
  for (int i = tid; i < 64 * 64; i += 128) {
    int c = i >> 6, p = i & 63;
    sA[p * 64 + c] = (f16)xb[(size_t)c * HWp + p0 + p];
  }
  for (int i = tid; i < 96 * 64; i += 128) {
    int n = i >> 6, c = i & 63;
    float wv = (n < 64) ? Wa[n * 64 + c] : W1[(n - 64) * 64 + c];
    sW[n * 64 + c] = (f16)wv;
  }
  __syncthreads();

  const int wv   = tid >> 5, lane = tid & 31;
  const int lrow = lane & 15, lhi = lane >> 4;
  f32x8 acc[6] = {};

#pragma unroll
  for (int k0 = 0; k0 < 64; k0 += 32) {
    h16 afrag = *(const h16*)&sA[(16 * wv + lrow) * 64 + k0 + lhi * 16];
#pragma unroll
    for (int nt = 0; nt < 6; ++nt) {
      h16 bfrag = *(const h16*)&sW[(nt * 16 + lrow) * 64 + k0 + lhi * 16];
      acc[nt] = wmma_f16(afrag, bfrag, acc[nt]);
    }
  }

  const float al_a = aa[0], al_1 = a1[0];
#pragma unroll
  for (int nt = 0; nt < 6; ++nt) {
    int n = nt * 16 + lrow;
#pragma unroll
    for (int v = 0; v < 8; ++v) {
      int p = p0 + 16 * wv + v + 8 * lhi;     // C-matrix VGPR layout
      float val = acc[nt][v];
      if (n < 64) {
        val += ba[n];
        val = val >= 0.f ? val : al_a * val;
        embed[((size_t)b * 64 + n) * HWp + p] = (f16)val;
      } else {
        val += b1[n - 64];
        val = val >= 0.f ? val : al_1 * val;
        match[((size_t)b * 32 + (n - 64)) * HWp + p] = (f16)val;
      }
    }
  }
}

// ---------------------------------------------------------------------------
// 2. 2x2 mean pool + conv1x1 (W2) + PReLU -> ref f32 [B,32,48,48]
// ---------------------------------------------------------------------------
__global__ __launch_bounds__(256)
void k_pool_ref(const float* __restrict__ x, const float* __restrict__ W2,
                const float* __restrict__ b2, const float* __restrict__ a2,
                float* __restrict__ ref) {
  int idx = blockIdx.x * blockDim.x + threadIdx.x;
  if (idx >= BB * 32 * 2304) return;
  int p = idx % 2304;
  int co = (idx / 2304) % 32;
  int b  = idx / (2304 * 32);
  int pi = p / 48, pj = p % 48;
  const float* xb   = x + (size_t)b * CC * HWp;
  const float* wrow = W2 + co * 64;
  float acc = b2[co];
  int base = (2 * pi) * 96 + 2 * pj;
  for (int c = 0; c < 64; ++c) {
    const float* xc = xb + (size_t)c * HWp + base;
    float pool = 0.25f * (xc[0] + xc[1] + xc[96] + xc[97]);
    acc = fmaf(wrow[c], pool, acc);
  }
  float al = a2[0];
  acc = acc >= 0.f ? acc : al * acc;
  ref[((size_t)b * 32 + co) * 2304 + p] = acc;
}

// ---------------------------------------------------------------------------
// 3. Normalized 3x3 patches of ref -> wn f16 [B, L, 288]; one wave per l
// ---------------------------------------------------------------------------
__global__ __launch_bounds__(128)
void k_wn(const float* __restrict__ ref, f16* __restrict__ wn) {
  int gw   = (blockIdx.x * blockDim.x + threadIdx.x) >> 5;
  int lane = threadIdx.x & 31;
  if (gw >= BB * LL) return;
  int b = gw / LL, l = gw % LL;
  int pi = l / 48, pj = l % 48;
  const float* rb = ref + (size_t)b * 32 * 2304;
  float vals[9];
  float ssq = 0.f;
#pragma unroll
  for (int e = 0; e < 9; ++e) {
    int c = lane;                 // idx = lane*9+e -> c = lane, tap = e
    int kh = e / 3, kw = e % 3;
    int ri = pi - 1 + kh, rj = pj - 1 + kw;
    float v = 0.f;
    if (ri >= 0 && ri < 48 && rj >= 0 && rj < 48)
      v = rb[c * 2304 + ri * 48 + rj];
    vals[e] = v;
    ssq = fmaf(v, v, ssq);
  }
  for (int off = 16; off > 0; off >>= 1) ssq += __shfl_xor(ssq, off, 32);
  float norm = sqrtf(ssq);
  norm = norm > 1e-4f ? norm : 1e-4f;
  float inv = 1.f / norm;
  f16* out = wn + ((size_t)b * LL + l) * K288;
#pragma unroll
  for (int e = 0; e < 9; ++e) out[lane * 9 + e] = (f16)(vals[e] * inv);
}

// ---------------------------------------------------------------------------
// 4. 3x3 patches of match_in -> xp f16 [B, Q, 288]
// ---------------------------------------------------------------------------
__global__ __launch_bounds__(256)
void k_xp(const f16* __restrict__ match, f16* __restrict__ xp) {
  size_t idx = (size_t)blockIdx.x * blockDim.x + threadIdx.x;
  const size_t total = (size_t)BB * HWp * K288;
  if (idx >= total) return;
  int e = (int)(idx % K288);
  size_t t = idx / K288;
  int q = (int)(t % HWp);
  int b = (int)(t / HWp);
  int c = e / 9, r = e % 9, kh = r / 3, kw = r % 3;
  int u = q / 96, v = q % 96;
  int ri = u - 1 + kh, rj = v - 1 + kw;
  f16 val = (f16)0.f;
  if (ri >= 0 && ri < 96 && rj >= 0 && rj < 96)
    val = match[((size_t)b * 32 + c) * HWp + ri * 96 + rj];
  xp[idx] = val;
}

// ---------------------------------------------------------------------------
// 5. 6x6 stride-2 patches of embed -> rawW f16 [B, L, 64*36]
// ---------------------------------------------------------------------------
__global__ __launch_bounds__(256)
void k_rawW(const f16* __restrict__ embed, f16* __restrict__ rawW) {
  size_t idx = (size_t)blockIdx.x * blockDim.x + threadIdx.x;
  const size_t total = (size_t)BB * LL * NDYN;
  if (idx >= total) return;
  int n = (int)(idx % NDYN);
  size_t t = idx / NDYN;
  int l = (int)(t % LL);
  int b = (int)(t / LL);
  int o = n / 36, r = n % 36, kh = r / 6, kw = r % 6;
  int pi = l / 48, pj = l % 48;
  int ri = 2 * pi - 2 + kh, rj = 2 * pj - 2 + kw;
  f16 val = (f16)0.f;
  if (ri >= 0 && ri < 96 && rj >= 0 && rj < 96)
    val = embed[((size_t)b * 64 + o) * HWp + ri * 96 + rj];
  rawW[idx] = val;
}

// ---------------------------------------------------------------------------
// 7. Softmax over L per query: store exp(10*(s-max)) f16 + rowsum f32.
//    (1/rowsum is applied in the tconv GEMM epilogue.)
// ---------------------------------------------------------------------------
__global__ __launch_bounds__(256)
void k_softmax(const float* __restrict__ scores, f16* __restrict__ eprob,
               float* __restrict__ rowsum) {
  int gw   = (blockIdx.x * blockDim.x + threadIdx.x) >> 5;
  int lane = threadIdx.x & 31;
  if (gw >= BB * HWp) return;
  const float* row = scores + (size_t)gw * LL;
  float m = -3.0e38f;
  for (int i = lane; i < LL; i += 32) m = fmaxf(m, row[i]);
  for (int off = 16; off > 0; off >>= 1) m = fmaxf(m, __shfl_xor(m, off, 32));
  f16* er = eprob + (size_t)gw * LL;
  float s = 0.f;
  for (int i = lane; i < LL; i += 32) {
    float e = __expf(10.0f * (row[i] - m));
    er[i] = (f16)e;
    s += e;
  }
  for (int off = 16; off > 0; off >>= 1) s += __shfl_xor(s, off, 32);
  if (lane == 0) rowsum[gw] = s;
}

// ---------------------------------------------------------------------------
// 9. Overlap-add (gather form): yt[b,o,u,v] = sum of <=9 patch taps
// ---------------------------------------------------------------------------
__global__ __launch_bounds__(256)
void k_yt(const float* __restrict__ P, float* __restrict__ yt) {
  size_t idx = (size_t)blockIdx.x * blockDim.x + threadIdx.x;
  const size_t total = (size_t)BB * 64 * 192 * 192;
  if (idx >= total) return;
  int v = (int)(idx % 192);
  size_t t = idx / 192;
  int u = (int)(t % 192); t /= 192;
  int o = (int)(t % 64);
  int b = (int)(t / 64);
  const float* Pb = P + (size_t)b * HWp * (size_t)NDYN;
  float acc = 0.f;
#pragma unroll
  for (int di = 0; di < 3; ++di) {
    int i = (u >> 1) - 1 + di;
    if (i < 0 || i >= 96) continue;
    int a = u - 2 * i + 2;  // in [0,5]
#pragma unroll
    for (int dj = 0; dj < 3; ++dj) {
      int j = (v >> 1) - 1 + dj;
      if (j < 0 || j >= 96) continue;
      int bt = v - 2 * j + 2;
      acc += Pb[(size_t)(i * 96 + j) * NDYN + o * 36 + a * 6 + bt];
    }
  }
  yt[idx] = acc;
}

// ---------------------------------------------------------------------------
// 10. 3x3 stride-2 down-conv + bias, * (1/6) -> d_out [2,64,96,96] f32
// ---------------------------------------------------------------------------
__global__ __launch_bounds__(256)
void k_down(const float* __restrict__ yt, const float* __restrict__ Wd,
            const float* __restrict__ bd, float* __restrict__ out) {
  size_t idx = (size_t)blockIdx.x * blockDim.x + threadIdx.x;
  const size_t total = (size_t)BB * 64 * HWp;
  if (idx >= total) return;
  int p = (int)(idx % HWp);
  size_t t = idx / HWp;
  int oo = (int)(t % 64);
  int b  = (int)(t / 64);
  int i = p / 96, j = p % 96;
  const float* ytb = yt + (size_t)b * 64 * 192 * 192;
  float acc = bd[oo];
  for (int c = 0; c < 64; ++c) {
    const float* yc   = ytb + (size_t)c * 192 * 192;
    const float* wrow = Wd + ((size_t)oo * 64 + c) * 9;
#pragma unroll
    for (int dh = 0; dh < 3; ++dh) {
      int u = 2 * i + dh - 1;
      if (u < 0 || u >= 192) continue;
#pragma unroll
      for (int dw = 0; dw < 3; ++dw) {
        int v = 2 * j + dw - 1;
        if (v < 0 || v >= 192) continue;
        acc = fmaf(wrow[dh * 3 + dw], yc[u * 192 + v], acc);
      }
    }
  }
  out[idx] = acc * (1.f / 6.f);
}

// ---------------------------------------------------------------------------
extern "C" void kernel_launch(void* const* d_in, const int* in_sizes, int n_in,
                              void* d_out, int out_size, void* d_ws,
                              size_t ws_size, hipStream_t stream) {
  (void)in_sizes; (void)n_in; (void)out_size; (void)ws_size;
  const float* x  = (const float*)d_in[0];
  const float* W1 = (const float*)d_in[1];
  const float* b1 = (const float*)d_in[2];
  const float* a1 = (const float*)d_in[3];
  const float* W2 = (const float*)d_in[4];
  const float* b2 = (const float*)d_in[5];
  const float* a2 = (const float*)d_in[6];
  const float* Wa = (const float*)d_in[7];
  const float* ba = (const float*)d_in[8];
  const float* aa = (const float*)d_in[9];
  const float* Wd = (const float*)d_in[10];
  const float* bd = (const float*)d_in[11];

  // Workspace carve-out (~313 MB total).
  char* wsp = (char*)d_ws;
  size_t off = 0;
  auto alloc = [&](size_t bytes) -> void* {
    void* p = wsp + off;
    off = (off + bytes + 255) & ~(size_t)255;
    return p;
  };
  f16*   embed  = (f16*)alloc((size_t)BB * 64 * HWp * sizeof(f16));
  f16*   match  = (f16*)alloc((size_t)BB * 32 * HWp * sizeof(f16));
  float* refb   = (float*)alloc((size_t)BB * 32 * 2304 * sizeof(float));
  f16*   wn     = (f16*)alloc((size_t)BB * LL * K288 * sizeof(f16));
  f16*   xp     = (f16*)alloc((size_t)BB * HWp * K288 * sizeof(f16));
  f16*   rawW   = (f16*)alloc((size_t)BB * LL * NDYN * sizeof(f16));
  float* scores = (float*)alloc((size_t)BB * HWp * LL * sizeof(float));
  f16*   eprob  = (f16*)alloc((size_t)BB * HWp * LL * sizeof(f16));
  float* rowsum = (float*)alloc((size_t)BB * HWp * sizeof(float));
  float* ytbuf  = (float*)alloc((size_t)BB * 64 * 192 * 192 * sizeof(float));
  float* Pbuf   = scores;  // overlay: scores dead after k_softmax

  k_embed_match<<<dim3(HWp / 64, BB), 128, 0, stream>>>(
      x, Wa, ba, aa, W1, b1, a1, embed, match);
  k_pool_ref<<<(BB * 32 * 2304 + 255) / 256, 256, 0, stream>>>(
      x, W2, b2, a2, refb);
  k_wn<<<(BB * LL * 32 + 127) / 128, 128, 0, stream>>>(refb, wn);
  {
    size_t tot = (size_t)BB * HWp * K288;
    k_xp<<<(unsigned)((tot + 255) / 256), 256, 0, stream>>>(match, xp);
  }
  {
    size_t tot = (size_t)BB * LL * NDYN;
    k_rawW<<<(unsigned)((tot + 255) / 256), 256, 0, stream>>>(embed, rawW);
  }
  k_attn<<<dim3(HWp / 64, LL / 128, BB), 128, 0, stream>>>(xp, wn, scores);
  k_softmax<<<(BB * HWp * 32 + 255) / 256, 256, 0, stream>>>(
      scores, eprob, rowsum);
  k_tconv<<<dim3(HWp / 64, NDYN / 128, BB), 128, 0, stream>>>(
      eprob, rawW, rowsum, Pbuf);
  {
    size_t tot = (size_t)BB * 64 * 192 * 192;
    k_yt<<<(unsigned)((tot + 255) / 256), 256, 0, stream>>>(Pbuf, ytbuf);
  }
  {
    size_t tot = (size_t)BB * 64 * HWp;
    k_down<<<(unsigned)((tot + 255) / 256), 256, 0, stream>>>(
        ytbuf, Wd, bd, (float*)d_out);
  }
}
